// Multi3GruUser_81449759801693
// MI455X (gfx1250) — matrix operand order repaired and checked by reading, not compiled, not run on hardware
//
#include <hip/hip_runtime.h>
#include <hip/hip_bf16.h>
#include <math.h>

// ---------------------------------------------------------------------------
// Problem constants
// ---------------------------------------------------------------------------
#define RR 128
#define SS 32
#define WW 48
#define EMB 300
#define EMBP 320      // EMB padded to multiple of 32 (WMMA K)
#define HIDN 512
#define UF 20
#define DINR 532      // HID + UF
#define DINRP 544     // padded to multiple of 32
#define NSENT (RR * SS)           // 4096
#define NTOK (RR * SS * WW)       // 196608

#define MB 32         // batch rows per workgroup (2 WMMA M-tiles)
#define NWAVE 16      // 512 threads
#define CT 2          // column tiles (of 16 hidden units) per wave: 16*2*16 = 512

typedef __attribute__((ext_vector_type(16))) __bf16 bf16x16;
typedef __attribute__((ext_vector_type(8)))  float  v8f;

union Frag {
  uint4   u4[2];
  bf16x16 v;
};

__device__ __forceinline__ bf16x16 frag_2x16B(const void* p0, const void* p1) {
  Frag f;
  f.u4[0] = *reinterpret_cast<const uint4*>(p0);
  f.u4[1] = *reinterpret_cast<const uint4*>(p1);
  return f.v;
}

__device__ __forceinline__ v8f wmma_bf16(bf16x16 a, bf16x16 b, v8f c) {
  // D = A(16x32 bf16) * B(32x16 bf16) + C(16x16 f32)
  return __builtin_amdgcn_wmma_f32_16x16x32_bf16(false, a, false, b,
                                                 (short)0, c, false, false);
}

// ---------------------------------------------------------------------------
// Bidirectional GRU recurrence, WMMA-based.
// One block = 32 batch rows (2 M-tiles); blockIdx.y selects direction.
// Carried f32 hidden state lives in registers (each lane owns its C/D
// elements); only the bf16 copy (WMMA A operand for h*Whh^T) is in LDS.
// x:   [batch][T][Din] bf16 (Din padded, mult of 32)
// Wih: [3*HIDN][Din] bf16, Whh: [3*HIDN][HIDN] bf16 (row-major, padded K)
// hout: [batch][HIDN] f32 (final hidden state per direction)
// ---------------------------------------------------------------------------
__global__ __launch_bounds__(512) void gru_wmma(
    const __hip_bfloat16* __restrict__ x, int T, int Din, int sentStride,
    const int* __restrict__ lens,
    const __hip_bfloat16* __restrict__ WihF, const __hip_bfloat16* __restrict__ WhhF,
    const float* __restrict__ bihF, const float* __restrict__ bhhF,
    const __hip_bfloat16* __restrict__ WihB, const __hip_bfloat16* __restrict__ WhhB,
    const float* __restrict__ bihB, const float* __restrict__ bhhB,
    float* __restrict__ houtF, float* __restrict__ houtB)
{
  const int dir = blockIdx.y;
  const __hip_bfloat16* Wih = dir ? WihB : WihF;
  const __hip_bfloat16* Whh = dir ? WhhB : WhhF;
  const float* bih = dir ? bihB : bihF;
  const float* bhh = dir ? bhhB : bhhF;
  float* hout = dir ? houtB : houtF;

  __shared__ __align__(16) __hip_bfloat16 hB16[MB * HIDN];  // bf16 h (WMMA A)

  const int tid = threadIdx.x;
  for (int i = tid; i < MB * HIDN; i += 512) hB16[i] = __float2bfloat16(0.0f);

  const int wave = tid >> 5;
  const int lane = tid & 31;
  const int nn = lane & 15;   // A row (M) / B+C column (N) within tile
  const int hi = lane >> 4;   // lane-half selector
  const int sbase = blockIdx.x * MB;

  int lenRow[2][8];
#pragma unroll
  for (int m = 0; m < 2; ++m)
#pragma unroll
    for (int v = 0; v < 8; ++v)
      lenRow[m][v] = lens[sbase + m * 16 + v + 8 * hi];

  float biasR[CT], biasZ[CT], biasGN[CT], biasHN[CT];
#pragma unroll
  for (int c = 0; c < CT; ++c) {
    const int j = (wave * CT + c) * 16 + nn;
    biasR[c]  = bih[j] + bhh[j];
    biasZ[c]  = bih[HIDN + j] + bhh[HIDN + j];
    biasGN[c] = bih[2 * HIDN + j];   // gi_n bias (n-gate needs r*gh_n separately)
    biasHN[c] = bhh[2 * HIDN + j];   // gh_n bias
  }

  // carried f32 hidden state, register-resident: hReg[m][c][v]
  float hReg[2][CT][8];
#pragma unroll
  for (int m = 0; m < 2; ++m)
#pragma unroll
    for (int c = 0; c < CT; ++c)
#pragma unroll
      for (int v = 0; v < 8; ++v) hReg[m][c][v] = 0.0f;

  __syncthreads();

  for (int s = 0; s < T; ++s) {
    const int t = dir ? (T - 1 - s) : s;

    v8f accR[2][CT], accZ[2][CT], accGN[2][CT], accHN[2][CT];
#pragma unroll
    for (int m = 0; m < 2; ++m)
#pragma unroll
      for (int c = 0; c < CT; ++c)
#pragma unroll
        for (int v = 0; v < 8; ++v) {
          accR[m][c][v]  = biasR[c];
          accZ[m][c][v]  = biasZ[c];
          accGN[m][c][v] = biasGN[c];
          accHN[m][c][v] = biasHN[c];
        }

    // -------- gi = x_t * Wih^T  (A from global bf16 input) --------
    const __hip_bfloat16* xrow0 =
        x + (size_t)(sbase + nn) * sentStride + (size_t)t * Din + hi * 8;
    const __hip_bfloat16* xrow1 =
        x + (size_t)(sbase + 16 + nn) * sentStride + (size_t)t * Din + hi * 8;
    for (int k0 = 0; k0 < Din; k0 += 32) {
      const bf16x16 a0 = frag_2x16B(xrow0 + k0, xrow0 + k0 + 16);
      const bf16x16 a1 = frag_2x16B(xrow1 + k0, xrow1 + k0 + 16);
#pragma unroll
      for (int c = 0; c < CT; ++c) {
        const size_t wrow = (size_t)((wave * CT + c) * 16 + nn);
        const size_t gstride = (size_t)HIDN * Din;
        const __hip_bfloat16* wp = Wih + wrow * Din + k0 + hi * 8;
        const bf16x16 bR = frag_2x16B(wp, wp + 16);
        const bf16x16 bZ = frag_2x16B(wp + gstride, wp + gstride + 16);
        const bf16x16 bN = frag_2x16B(wp + 2 * gstride, wp + 2 * gstride + 16);
        accR[0][c]  = wmma_bf16(a0, bR, accR[0][c]);
        accR[1][c]  = wmma_bf16(a1, bR, accR[1][c]);
        accZ[0][c]  = wmma_bf16(a0, bZ, accZ[0][c]);
        accZ[1][c]  = wmma_bf16(a1, bZ, accZ[1][c]);
        accGN[0][c] = wmma_bf16(a0, bN, accGN[0][c]);
        accGN[1][c] = wmma_bf16(a1, bN, accGN[1][c]);
      }
    }

    // -------- gh = h * Whh^T  (A from LDS bf16 hidden state) --------
    const __hip_bfloat16* hrow0 = &hB16[nn * HIDN + hi * 8];
    const __hip_bfloat16* hrow1 = &hB16[(16 + nn) * HIDN + hi * 8];
    for (int k0 = 0; k0 < HIDN; k0 += 32) {
      const bf16x16 a0 = frag_2x16B(hrow0 + k0, hrow0 + k0 + 16);
      const bf16x16 a1 = frag_2x16B(hrow1 + k0, hrow1 + k0 + 16);
#pragma unroll
      for (int c = 0; c < CT; ++c) {
        const size_t wrow = (size_t)((wave * CT + c) * 16 + nn);
        const size_t gstride = (size_t)HIDN * HIDN;
        const __hip_bfloat16* wp = Whh + wrow * HIDN + k0 + hi * 8;
        const bf16x16 bR = frag_2x16B(wp, wp + 16);
        const bf16x16 bZ = frag_2x16B(wp + gstride, wp + gstride + 16);
        const bf16x16 bN = frag_2x16B(wp + 2 * gstride, wp + 2 * gstride + 16);
        accR[0][c]  = wmma_bf16(a0, bR, accR[0][c]);
        accR[1][c]  = wmma_bf16(a1, bR, accR[1][c]);
        accZ[0][c]  = wmma_bf16(a0, bZ, accZ[0][c]);
        accZ[1][c]  = wmma_bf16(a1, bZ, accZ[1][c]);
        accHN[0][c] = wmma_bf16(a0, bN, accHN[0][c]);
        accHN[1][c] = wmma_bf16(a1, bN, accHN[1][c]);
      }
    }

    __syncthreads();  // all waves finished reading hB16 for this step

    // -------- gates + masked hidden update (h carried in registers) --------
#pragma unroll
    for (int m = 0; m < 2; ++m)
#pragma unroll
      for (int c = 0; c < CT; ++c) {
        const int j = (wave * CT + c) * 16 + nn;
#pragma unroll
        for (int v = 0; v < 8; ++v) {
          const float rg = 1.0f / (1.0f + __expf(-accR[m][c][v]));
          const float zg = 1.0f / (1.0f + __expf(-accZ[m][c][v]));
          const float ng = tanhf(accGN[m][c][v] + rg * accHN[m][c][v]);
          const float hOld = hReg[m][c][v];
          const float hNew = (t < lenRow[m][v])
                                 ? ((1.0f - zg) * ng + zg * hOld)
                                 : hOld;
          hReg[m][c][v] = hNew;
          const int row = m * 16 + v + 8 * hi;  // C/D layout: M = vgpr + 8*(lane>=16)
          hB16[row * HIDN + j] = __float2bfloat16(hNew);
        }
      }
    __syncthreads();  // writes visible before next step's A fragments
  }

  // final hidden state from registers
#pragma unroll
  for (int m = 0; m < 2; ++m)
#pragma unroll
    for (int c = 0; c < CT; ++c) {
      const int j = (wave * CT + c) * 16 + nn;
#pragma unroll
      for (int v = 0; v < 8; ++v) {
        const int row = m * 16 + v + 8 * hi;
        hout[(size_t)(sbase + row) * HIDN + j] = hReg[m][c][v];
      }
    }
}

// ---------------------------------------------------------------------------
// Embedding gather -> bf16, K padded 300 -> 320
// ---------------------------------------------------------------------------
__global__ void gather_embed(const int* __restrict__ tok,
                             const float* __restrict__ embed,
                             __hip_bfloat16* __restrict__ x) {
  const long idx = (long)blockIdx.x * 256 + threadIdx.x;
  const long total = (long)NTOK * (EMBP / 4);
  if (idx >= total) return;
  const long tokIdx = idx / (EMBP / 4);
  const int col = (int)(idx % (EMBP / 4)) * 4;
  union { __hip_bfloat16 h[4]; uint2 u; } o;
  if (col < EMB) {
    const float4 v =
        *reinterpret_cast<const float4*>(embed + (size_t)tok[tokIdx] * EMB + col);
    o.h[0] = __float2bfloat16(v.x);
    o.h[1] = __float2bfloat16(v.y);
    o.h[2] = __float2bfloat16(v.z);
    o.h[3] = __float2bfloat16(v.w);
  } else {
    o.h[0] = o.h[1] = o.h[2] = o.h[3] = __float2bfloat16(0.0f);
  }
  *reinterpret_cast<uint2*>(x + tokIdx * EMBP + col) = o.u;
}

// f32 weights -> bf16, zero-padded along K (kin -> kout)
__global__ void convert_pad(const float* __restrict__ in,
                            __hip_bfloat16* __restrict__ out,
                            int rows, int kin, int kout) {
  const long idx = (long)blockIdx.x * 256 + threadIdx.x;
  if (idx >= (long)rows * kout) return;
  const int r = (int)(idx / kout);
  const int k = (int)(idx % kout);
  out[idx] = (k < kin) ? __float2bfloat16(in[(size_t)r * kin + k])
                       : __float2bfloat16(0.0f);
}

// bf16(a[i] + b[i]) -- builds sentence-level input (hf + hb)
__global__ void sum_to_bf16(const float* __restrict__ a, const float* __restrict__ b,
                            __hip_bfloat16* __restrict__ o, long n) {
  const long i = (long)blockIdx.x * 256 + threadIdx.x;
  if (i < n) o[i] = __float2bfloat16(a[i] + b[i]);
}

// zero review-level input (32 padded batch rows) + lens vector {128,0,...}
__global__ void init_review_aux(__hip_bfloat16* __restrict__ xr,
                                int* __restrict__ lensr) {
  const long i = (long)blockIdx.x * 256 + threadIdx.x;
  if (i < (long)MB * RR * DINRP) xr[i] = __float2bfloat16(0.0f);
  if (i < MB) lensr[i] = (i == 0) ? RR : 0;
}

// p_batch (f32, [R][532]) and review-GRU input row 0 (bf16, [R][544])
__global__ void build_review_input(const float* __restrict__ hf,
                                   const float* __restrict__ hb,
                                   const float* __restrict__ uf,
                                   float* __restrict__ pbatch,
                                   __hip_bfloat16* __restrict__ xr) {
  const int r = blockIdx.x;
  __shared__ float nrm;
  if (threadIdx.x == 0) {
    float ss = 0.0f;
    for (int i = 0; i < UF; ++i) { float u = uf[r * UF + i]; ss += u * u; }
    nrm = fmaxf(sqrtf(ss), 1e-12f);
  }
  __syncthreads();
  for (int k = threadIdx.x; k < DINRP; k += 256) {
    float v;
    if (k < HIDN)       v = hf[(size_t)r * HIDN + k] + hb[(size_t)r * HIDN + k];
    else if (k < DINR)  v = uf[r * UF + (k - HIDN)] / nrm;
    else                v = 0.0f;
    if (k < DINR) pbatch[(size_t)r * DINR + k] = v;
    xr[(size_t)r * DINRP + k] = __float2bfloat16(v);  // batch row 0 of [MB][R][544]
  }
}

__device__ __forceinline__ float selu_f(float x) {
  const float scale = 1.0507009873554805f, alpha = 1.6732632423543772f;
  return scale * (x > 0.0f ? x : alpha * (__expf(x) - 1.0f));
}

// r_fc: per-review MLP 532 -> 256 (selu) -> 1
__global__ void mlp_rfc(const float* __restrict__ pbatch,
                        const float* __restrict__ W1, const float* __restrict__ b1,
                        const float* __restrict__ W2, const float* __restrict__ b2,
                        float* __restrict__ out) {
  const int r = blockIdx.x, j = threadIdx.x;
  const float* xrow = pbatch + (size_t)r * DINR;
  float acc = b1[j];
  for (int k = 0; k < DINR; ++k) acc += xrow[k] * W1[(size_t)j * DINR + k];
  __shared__ float red[256];
  red[j] = selu_f(acc) * W2[j];
  __syncthreads();
  for (int sft = 128; sft > 0; sft >>= 1) {
    if (j < sft) red[j] += red[j + sft];
    __syncthreads();
  }
  if (j == 0) out[r] = red[0] + b2[0];
}

// p_fc: 512 -> 256 (selu) -> 1 on h_n = hf[0]+hb[0]
__global__ void mlp_pfc(const float* __restrict__ hf, const float* __restrict__ hb,
                        const float* __restrict__ W1, const float* __restrict__ b1,
                        const float* __restrict__ W2, const float* __restrict__ b2,
                        float* __restrict__ out) {
  const int j = threadIdx.x;
  float acc = b1[j];
  for (int k = 0; k < HIDN; ++k) acc += (hf[k] + hb[k]) * W1[(size_t)j * HIDN + k];
  __shared__ float red[256];
  red[j] = selu_f(acc) * W2[j];
  __syncthreads();
  for (int sft = 128; sft > 0; sft >>= 1) {
    if (j < sft) red[j] += red[j + sft];
    __syncthreads();
  }
  if (j == 0) out[0] = red[0] + b2[0];
}

// ---------------------------------------------------------------------------
// Host launcher
// ---------------------------------------------------------------------------
extern "C" void kernel_launch(void* const* d_in, const int* in_sizes, int n_in,
                              void* d_out, int out_size, void* d_ws, size_t ws_size,
                              hipStream_t stream) {
  (void)in_sizes; (void)n_in; (void)out_size; (void)ws_size;
  // Input order: jax-pytree (sorted-key) flatten of setup_inputs():
  //  0 inputs 1 sent_lengths 2 sent_counts 3 user_feats 4 embed
  //  params: p_fc{5:W1 6:W2 7:b1 8:b2}
  //          r{9:Whh_b 10:Whh_f 11:Wih_b 12:Wih_f 13:bhh_b 14:bhh_f 15:bih_b 16:bih_f}
  //          r_fc{17:W1 18:W2 19:b1 20:b2}
  //          s{21:Whh_b 22:Whh_f 23:Wih_b 24:Wih_f 25:bhh_b 26:bhh_f 27:bih_b 28:bih_f}
  //          w{29:Whh_b 30:Whh_f 31:Wih_b 32:Wih_f 33:bhh_b 34:bhh_f 35:bih_b 36:bih_f}
  const int*   tok   = (const int*)d_in[0];
  const int*   slen  = (const int*)d_in[1];
  const int*   scnt  = (const int*)d_in[2];
  const float* uf    = (const float*)d_in[3];
  const float* embed = (const float*)d_in[4];
  float* out = (float*)d_out;

  char* ws = (char*)d_ws;
  size_t off = 0;
  auto alloc = [&](size_t bytes) -> char* {
    char* p = ws + off;
    off = (off + bytes + 255) & ~(size_t)255;
    return p;
  };
  typedef __hip_bfloat16 bf;
  bf* xw      = (bf*)alloc((size_t)NTOK * EMBP * 2);
  bf* wihw_f  = (bf*)alloc((size_t)3 * HIDN * EMBP * 2);
  bf* wihw_b  = (bf*)alloc((size_t)3 * HIDN * EMBP * 2);
  bf* whhw_f  = (bf*)alloc((size_t)3 * HIDN * HIDN * 2);
  bf* whhw_b  = (bf*)alloc((size_t)3 * HIDN * HIDN * 2);
  bf* wihs_f  = (bf*)alloc((size_t)3 * HIDN * HIDN * 2);
  bf* wihs_b  = (bf*)alloc((size_t)3 * HIDN * HIDN * 2);
  bf* whhs_f  = (bf*)alloc((size_t)3 * HIDN * HIDN * 2);
  bf* whhs_b  = (bf*)alloc((size_t)3 * HIDN * HIDN * 2);
  bf* wihr_f  = (bf*)alloc((size_t)3 * HIDN * DINRP * 2);
  bf* wihr_b  = (bf*)alloc((size_t)3 * HIDN * DINRP * 2);
  bf* whhr_f  = (bf*)alloc((size_t)3 * HIDN * HIDN * 2);
  bf* whhr_b  = (bf*)alloc((size_t)3 * HIDN * HIDN * 2);
  float* senthf = (float*)alloc((size_t)NSENT * HIDN * 4);
  float* senthb = (float*)alloc((size_t)NSENT * HIDN * 4);
  bf*    xs     = (bf*)alloc((size_t)NSENT * HIDN * 2);
  float* revhf  = (float*)alloc((size_t)RR * HIDN * 4);
  float* revhb  = (float*)alloc((size_t)RR * HIDN * 4);
  bf*    xr     = (bf*)alloc((size_t)MB * RR * DINRP * 2);
  float* pbatch = (float*)alloc((size_t)RR * DINR * 4);
  float* revrf  = (float*)alloc((size_t)MB * HIDN * 4);
  float* revrb  = (float*)alloc((size_t)MB * HIDN * 4);
  int*   lensr  = (int*)alloc(MB * 4);

  auto cgrid = [](long n) { return dim3((unsigned)((n + 255) / 256)); };

  // ---- weight conversions to bf16 (K padded to mult of 32) ----
  long nW;
  nW = (long)3 * HIDN * EMBP;
  convert_pad<<<cgrid(nW), 256, 0, stream>>>((const float*)d_in[32], wihw_f, 3 * HIDN, EMB, EMBP);
  convert_pad<<<cgrid(nW), 256, 0, stream>>>((const float*)d_in[31], wihw_b, 3 * HIDN, EMB, EMBP);
  nW = (long)3 * HIDN * HIDN;
  convert_pad<<<cgrid(nW), 256, 0, stream>>>((const float*)d_in[30], whhw_f, 3 * HIDN, HIDN, HIDN);
  convert_pad<<<cgrid(nW), 256, 0, stream>>>((const float*)d_in[29], whhw_b, 3 * HIDN, HIDN, HIDN);
  convert_pad<<<cgrid(nW), 256, 0, stream>>>((const float*)d_in[24], wihs_f, 3 * HIDN, HIDN, HIDN);
  convert_pad<<<cgrid(nW), 256, 0, stream>>>((const float*)d_in[23], wihs_b, 3 * HIDN, HIDN, HIDN);
  convert_pad<<<cgrid(nW), 256, 0, stream>>>((const float*)d_in[22], whhs_f, 3 * HIDN, HIDN, HIDN);
  convert_pad<<<cgrid(nW), 256, 0, stream>>>((const float*)d_in[21], whhs_b, 3 * HIDN, HIDN, HIDN);
  convert_pad<<<cgrid(nW), 256, 0, stream>>>((const float*)d_in[10], whhr_f, 3 * HIDN, HIDN, HIDN);
  convert_pad<<<cgrid(nW), 256, 0, stream>>>((const float*)d_in[9],  whhr_b, 3 * HIDN, HIDN, HIDN);
  nW = (long)3 * HIDN * DINRP;
  convert_pad<<<cgrid(nW), 256, 0, stream>>>((const float*)d_in[12], wihr_f, 3 * HIDN, DINR, DINRP);
  convert_pad<<<cgrid(nW), 256, 0, stream>>>((const float*)d_in[11], wihr_b, 3 * HIDN, DINR, DINRP);

  // ---- embedding gather ----
  gather_embed<<<cgrid((long)NTOK * (EMBP / 4)), 256, 0, stream>>>(tok, embed, xw);

  // ---- word-level BiGRU: 4096 sentences, T=48, Din=320 ----
  gru_wmma<<<dim3(NSENT / MB, 2), 512, 0, stream>>>(
      xw, WW, EMBP, WW * EMBP, slen,
      wihw_f, whhw_f, (const float*)d_in[36], (const float*)d_in[34],
      wihw_b, whhw_b, (const float*)d_in[35], (const float*)d_in[33],
      senthf, senthb);

  // ---- sentence-level input: bf16(hf + hb) ----
  sum_to_bf16<<<cgrid((long)NSENT * HIDN), 256, 0, stream>>>(
      senthf, senthb, xs, (long)NSENT * HIDN);

  // ---- sentence-level BiGRU: 128 reviews, T=32, Din=512 ----
  gru_wmma<<<dim3(RR / MB, 2), 512, 0, stream>>>(
      xs, SS, HIDN, SS * HIDN, scnt,
      wihs_f, whhs_f, (const float*)d_in[28], (const float*)d_in[26],
      wihs_b, whhs_b, (const float*)d_in[27], (const float*)d_in[25],
      revhf, revhb);

  // ---- review-level input (batch padded 1 -> 32 via len mask) ----
  init_review_aux<<<cgrid((long)MB * RR * DINRP), 256, 0, stream>>>(xr, lensr);
  build_review_input<<<dim3(RR), 256, 0, stream>>>(revhf, revhb, uf, pbatch, xr);

  // ---- review-level BiGRU: batch 32 (1 real), T=128, Din=544 ----
  gru_wmma<<<dim3(1, 2), 512, 0, stream>>>(
      xr, RR, DINRP, RR * DINRP, lensr,
      wihr_f, whhr_f, (const float*)d_in[16], (const float*)d_in[14],
      wihr_b, whhr_b, (const float*)d_in[15], (const float*)d_in[13],
      revrf, revrb);

  // ---- MLP heads ----
  mlp_rfc<<<dim3(RR), 256, 0, stream>>>(
      pbatch, (const float*)d_in[17], (const float*)d_in[19],
      (const float*)d_in[18], (const float*)d_in[20], out + 1);
  mlp_pfc<<<dim3(1), 256, 0, stream>>>(
      revrf, revrb, (const float*)d_in[5], (const float*)d_in[7],
      (const float*)d_in[6], (const float*)d_in[8], out);
}
